// UNetFlow_523986010281
// MI455X (gfx1250) — matrix operand order repaired and checked by reading, not compiled, not run on hardware
//
#include <hip/hip_runtime.h>
#include <hip/hip_bf16.h>

typedef __attribute__((ext_vector_type(16))) _Float16 v16h;
typedef __attribute__((ext_vector_type(8)))  _Float16 v8h;
typedef __attribute__((ext_vector_type(8)))  float    v8f;

#define BATCH   8
#define S_TOK   1024
#define N_COND  1024
#define DMODEL  256
#define RELD    64
#define NHEADS  4
#define DHEAD   64
#define KSEL    614
#define NEGBIG  -1e9f

// Combine two contiguous 8-half chunks into a 16-half WMMA fragment register.
__device__ __forceinline__ v16h ld16h(const _Float16* p0, const _Float16* p1)
{
    v8h lo = *(const v8h*)p0;
    v8h hi = *(const v8h*)p1;
    return __builtin_shufflevector(lo, hi, 0,1,2,3,4,5,6,7,8,9,10,11,12,13,14,15);
}

// ---------------------------------------------------------------------------
// Generic WMMA GEMM:  out[b][m][n] = sum_k A[b][m][k] * W[n][k] + bias[n]
// TA: input element type of A (float or _Float16); TO: output element type.
// A/out have arbitrary strides (read h transposed in place; write V transposed
// or the final (B,C,S) layout directly).  One wave = one 16x16 tile,
// K-loop in steps of 32 via v_wmma_f32_16x16x32_f16, f32 accumulation.
// ---------------------------------------------------------------------------
template <typename TA, typename TO>
__global__ __launch_bounds__(32) void gemm_abT_wmma(
    const TA* __restrict__ A, long long aBatch, long long aStrideM, long long aStrideK,
    const float* __restrict__ W, const float* __restrict__ bias, int Kdim,
    TO* __restrict__ out, long long oBatch, long long oStrideM, long long oStrideN)
{
    const int lane = threadIdx.x;
    const int l15  = lane & 15;
    const int hiA  = (lane & 16) ? 8 : 0;    // A-layout K offset for upper half-wave
    const int hiB  = (lane & 16) ? 16 : 0;   // K offset for upper half-wave (A and B both use k = 16h + s)
    const int tM   = blockIdx.x * 16;
    const int tN   = blockIdx.y * 16;
    const int b    = blockIdx.z;

    const TA*    Arow = A + (long long)b * aBatch + (long long)(tM + l15) * aStrideM;
    const float* Wrow = W + (long long)(tN + l15) * Kdim;

    v8f acc = {};
    for (int k0 = 0; k0 < Kdim; k0 += 32) {
        v16h a, bb;
#pragma unroll
        for (int j = 0; j < 8; ++j) {
            a[j]     = (_Float16)Arow[(long long)(k0 + hiB + j) * aStrideK];
            a[j + 8] = (_Float16)Arow[(long long)(k0 + hiB + 8 + j) * aStrideK];
        }
#pragma unroll
        for (int j = 0; j < 16; ++j)
            bb[j] = (_Float16)Wrow[k0 + hiB + j];
        acc = __builtin_amdgcn_wmma_f32_16x16x32_f16(false, a, false, bb,
                                                     (short)0, acc, false, false);
    }

    const int   col = tN + l15;
    const float bv  = bias ? bias[col] : 0.0f;
    TO* ob = out + (long long)b * oBatch;
    const int rowBase = tM + ((lane & 16) ? 8 : 0);
#pragma unroll
    for (int r = 0; r < 8; ++r)
        ob[(long long)(rowBase + r) * oStrideM + (long long)col * oStrideN]
            = (TO)(acc[r] + bv);
}

// ---------------------------------------------------------------------------
// L2-normalize rows of (rows, d) in place (eps matches reference 1e-12).
// ---------------------------------------------------------------------------
__global__ void l2norm_kernel(float* __restrict__ p, int rows, int d)
{
    int row = blockIdx.x * blockDim.x + threadIdx.x;
    if (row >= rows) return;
    float* r = p + (long long)row * d;
    float s = 0.0f;
    for (int i = 0; i < d; ++i) s += r[i] * r[i];
    float invn = 1.0f / fmaxf(sqrtf(s), 1e-12f);
    for (int i = 0; i < d; ++i) r[i] *= invn;
}

// colsum[b][j] = sum_n proj[b][n][j]      (block = 64 threads, grid = B)
__global__ void colsum_kernel(const float* __restrict__ p, float* __restrict__ cs)
{
    int b = blockIdx.x, j = threadIdx.x;
    const float* pb = p + (long long)b * N_COND * RELD;
    float s = 0.0f;
    for (int n = 0; n < N_COND; ++n) s += pb[(long long)n * RELD + j];
    cs[b * RELD + j] = s;
}

// cent[b][n] = dot(proj[b][n], colsum[b]) - dot(proj[b][n], proj[b][n])
__global__ void cent_kernel(const float* __restrict__ p, const float* __restrict__ cs,
                            float* __restrict__ cent)
{
    long long idx = (long long)blockIdx.x * blockDim.x + threadIdx.x;
    if (idx >= (long long)BATCH * N_COND) return;
    int b = (int)(idx / N_COND);
    const float* row = p + idx * RELD;
    const float* c   = cs + b * RELD;
    float dot = 0.0f, sq = 0.0f;
    for (int i = 0; i < RELD; ++i) { float v = row[i]; dot += v * c[i]; sq += v * v; }
    cent[idx] = dot - sq;
}

// ---------------------------------------------------------------------------
// Exact top-K_SEL selection via rank counting (tie-break: lower index wins,
// matching jax.lax.top_k), AND with cond_mask, argmax fallback if empty.
// Emits additive bias: 0 if allowed, -1e9 otherwise.  grid = B, block = 1024.
// ---------------------------------------------------------------------------
__global__ __launch_bounds__(1024) void select_kernel(
    const float* __restrict__ cent, const unsigned char* __restrict__ mask,
    float* __restrict__ maskbias)
{
    __shared__ float sc[N_COND];
    __shared__ int   sAny;
    const int b = blockIdx.x, n = threadIdx.x;
    if (n == 0) sAny = 0;
    sc[n] = cent[(long long)b * N_COND + n];
    __syncthreads();
    const float c = sc[n];
    int rank = 0;
    for (int m = 0; m < N_COND; ++m) {
        float cm = sc[m];
        rank += (cm > c) || (cm == c && m < n);
    }
    bool allowed = (rank < KSEL) && (mask[(long long)b * N_COND + n] != 0);
    if (allowed) atomicOr(&sAny, 1);
    __syncthreads();
    if (sAny == 0 && rank == 0) allowed = true;  // fallback to argmax(cent)
    maskbias[(long long)b * N_COND + n] = allowed ? 0.0f : NEGBIG;
}

// ---------------------------------------------------------------------------
// Flash cross-attention on pre-staged f16 operands, one wave per
// (b, head, 16-query tile).  Streams 32 keys/iter:
//   - Q/K fragments: vectorized b128 loads (layout-native f16 in memory)
//   - 4 WMMAs for scores (K=64), online softmax via 16-lane shfl_xor
//   - P transposed to A-layout through an f16 LDS tile (ds b128 readback)
//   - V^T rows give B-fragments as single contiguous 32B loads; 4 WMMAs for P.V
//   - prefetch of next K/V^T tiles (global_prefetch_b8)
// ---------------------------------------------------------------------------
__global__ __launch_bounds__(32) void flash_attn_kernel(
    const _Float16* __restrict__ Q,     // (B, S, 256) f16
    const _Float16* __restrict__ K,     // (B, N, 256) f16
    const _Float16* __restrict__ Vt,    // (B, 256, N) f16  (transposed V)
    const float* __restrict__ maskbias, // (B, N) f32
    _Float16* __restrict__ O)           // (B, S, 256) f16
{
    __shared__ _Float16 sP[16][40];     // 16x32 prob tile (+pad), f16
    const int lane = threadIdx.x;
    const int l15  = lane & 15;
    const int hiA  = (lane & 16) ? 8 : 0;
    const int hiB  = (lane & 16) ? 16 : 0;
    const int hiM  = (lane & 16) ? 8 : 0;   // C-layout row offset
    const int head = blockIdx.y;
    const int b    = blockIdx.z;
    const int s0   = blockIdx.x * 16;
    const float scale = 0.125f;             // 1/sqrt(64)

    // Q tile -> two A-layout fragments (16 rows x K=64), pure b128 loads
    const _Float16* qrow = Q + ((long long)b * S_TOK + (s0 + l15)) * DMODEL + head * DHEAD;
    const v16h aq0 = ld16h(qrow + hiB,      qrow + hiB + 8);
    const v16h aq1 = ld16h(qrow + 32 + hiB, qrow + 32 + hiB + 8);

    v8f oacc[4] = {};
    float rmax[8], rsum[8];
#pragma unroll
    for (int r = 0; r < 8; ++r) { rmax[r] = -3.0e38f; rsum[r] = 0.0f; }

    const _Float16* Kb  = K  + (long long)b * N_COND * DMODEL;
    const _Float16* Vtb = Vt + (long long)b * DMODEL * N_COND + (head * DHEAD) * N_COND;
    const float*    mb  = maskbias + (long long)b * N_COND;

    for (int kt = 0; kt < N_COND; kt += 32) {
        if (kt + 32 < N_COND) {   // prefetch next key/value tiles into cache
            __builtin_prefetch(Kb + (long long)(kt + 32 + l15) * DMODEL + head * DHEAD, 0, 3);
            __builtin_prefetch(Vtb + (long long)l15 * N_COND + kt + 32, 0, 3);
        }

        // ---- scores: two 16-key tiles, K=64 contraction (2 WMMAs each) ----
        v8f sc2[2];
#pragma unroll
        for (int t = 0; t < 2; ++t) {
            const _Float16* krow =
                Kb + (long long)(kt + t * 16 + l15) * DMODEL + head * DHEAD;
            const v16h bk0 = ld16h(krow + hiB,      krow + hiB + 8);
            const v16h bk1 = ld16h(krow + 32 + hiB, krow + 32 + hiB + 8);
            v8f c = {};
            c = __builtin_amdgcn_wmma_f32_16x16x32_f16(false, aq0, false, bk0,
                                                       (short)0, c, false, false);
            c = __builtin_amdgcn_wmma_f32_16x16x32_f16(false, aq1, false, bk1,
                                                       (short)0, c, false, false);
            const float add = mb[kt + t * 16 + l15];
#pragma unroll
            for (int r = 0; r < 8; ++r) sc2[t][r] = c[r] * scale + add;
        }

        // ---- online softmax per row (row r+hiM lives across 16 lanes) ----
#pragma unroll
        for (int r = 0; r < 8; ++r) {
            float m = fmaxf(sc2[0][r], sc2[1][r]);
#pragma unroll
            for (int off = 8; off >= 1; off >>= 1)
                m = fmaxf(m, __shfl_xor(m, off, 16));
            const float nm   = fmaxf(rmax[r], m);
            const float corr = __expf(rmax[r] - nm);
            const float p0   = __expf(sc2[0][r] - nm);
            const float p1   = __expf(sc2[1][r] - nm);
            float ps = p0 + p1;
#pragma unroll
            for (int off = 8; off >= 1; off >>= 1)
                ps += __shfl_xor(ps, off, 16);
            rsum[r] = rsum[r] * corr + ps;
            rmax[r] = nm;
#pragma unroll
            for (int t = 0; t < 4; ++t) oacc[t][r] *= corr;
            sc2[0][r] = p0;
            sc2[1][r] = p1;
        }

        // ---- transpose P (C-layout -> A-layout) through f16 LDS ----
        __syncthreads();
#pragma unroll
        for (int r = 0; r < 8; ++r) {
            sP[r + hiM][l15]      = (_Float16)sc2[0][r];
            sP[r + hiM][16 + l15] = (_Float16)sc2[1][r];
        }
        __syncthreads();
        const v16h ap = ld16h(&sP[l15][hiB], &sP[l15][hiB + 8]);

        // ---- out += P (16x32) @ V (32 x 64): V^T rows are contiguous ----
#pragma unroll
        for (int t = 0; t < 4; ++t) {
            const _Float16* vrow =
                Vtb + (long long)(t * 16 + l15) * N_COND + kt + hiB;
            const v16h bv = ld16h(vrow, vrow + 8);
            oacc[t] = __builtin_amdgcn_wmma_f32_16x16x32_f16(false, ap, false, bv,
                                                             (short)0, oacc[t], false, false);
        }
    }

    // ---- normalize and write (B, S, 256) f16; heads contiguous in a row ----
#pragma unroll
    for (int r = 0; r < 8; ++r) {
        const float inv = 1.0f / rsum[r];
        _Float16* orow =
            O + ((long long)b * S_TOK + (s0 + r + hiM)) * DMODEL + head * DHEAD + l15;
#pragma unroll
        for (int t = 0; t < 4; ++t) orow[t * 16] = (_Float16)(oacc[t][r] * inv);
    }
}

// ---------------------------------------------------------------------------
extern "C" void kernel_launch(void* const* d_in, const int* in_sizes, int n_in,
                              void* d_out, int out_size, void* d_ws, size_t ws_size,
                              hipStream_t stream)
{
    const float* h     = (const float*)d_in[0];   // (8,256,32,32)
    const float* cond  = (const float*)d_in[1];   // (8,1024,256)
    const unsigned char* mask = (const unsigned char*)d_in[2]; // (8,1024) bool
    const float* q_w = (const float*)d_in[3];
    const float* q_b = (const float*)d_in[4];
    const float* k_w = (const float*)d_in[5];
    const float* k_b = (const float*)d_in[6];
    const float* v_w = (const float*)d_in[7];
    const float* v_b = (const float*)d_in[8];
    // d_in[9], d_in[10]: rq_w/rq_b — dead in the reference output, skipped.
    const float* rk_w = (const float*)d_in[11];
    const float* rk_b = (const float*)d_in[12];
    const float* o_w  = (const float*)d_in[13];
    const float* o_b  = (const float*)d_in[14];
    float* out = (float*)d_out;                   // (8,256,32,32)

    // workspace carve-up (f16 staging buffers: 4 MB each)
    char* ws = (char*)d_ws;
    const size_t szHalf = (size_t)BATCH * S_TOK * DMODEL * sizeof(_Float16);
    _Float16* Qh  = (_Float16*)ws;            ws += szHalf;
    _Float16* Kh  = (_Float16*)ws;            ws += szHalf;
    _Float16* Vth = (_Float16*)ws;            ws += szHalf;   // (B, 256, N)
    _Float16* AOh = (_Float16*)ws;            ws += szHalf;
    float* Rp    = (float*)ws;                ws += (size_t)BATCH * N_COND * RELD * sizeof(float);
    float* Csum  = (float*)ws;                ws += (size_t)BATCH * RELD * sizeof(float);
    float* Cent  = (float*)ws;                ws += (size_t)BATCH * N_COND * sizeof(float);
    float* Mbias = (float*)ws;

    const long long hBatch = (long long)DMODEL * S_TOK;
    const long long cBatch = (long long)N_COND * DMODEL;

    // Q = h^T @ q_w^T + q_b  -> f16 (B,S,256)
    gemm_abT_wmma<float, _Float16><<<dim3(S_TOK/16, DMODEL/16, BATCH), 32, 0, stream>>>(
        h, hBatch, /*aM*/1, /*aK*/S_TOK, q_w, q_b, DMODEL,
        Qh, (long long)S_TOK*DMODEL, DMODEL, 1);
    // K = cond @ k_w^T + k_b -> f16 (B,N,256)
    gemm_abT_wmma<float, _Float16><<<dim3(N_COND/16, DMODEL/16, BATCH), 32, 0, stream>>>(
        cond, cBatch, DMODEL, 1, k_w, k_b, DMODEL,
        Kh, (long long)N_COND*DMODEL, DMODEL, 1);
    // V = cond @ v_w^T + v_b -> f16 stored TRANSPOSED (B,256,N) via out strides
    gemm_abT_wmma<float, _Float16><<<dim3(N_COND/16, DMODEL/16, BATCH), 32, 0, stream>>>(
        cond, cBatch, DMODEL, 1, v_w, v_b, DMODEL,
        Vth, (long long)DMODEL*N_COND, /*oM(key)*/1, /*oN(dim)*/N_COND);
    // Rp = cond @ rk_w^T + rk_b -> f32 (selector path stays f32)
    gemm_abT_wmma<float, float><<<dim3(N_COND/16, RELD/16, BATCH), 32, 0, stream>>>(
        cond, cBatch, DMODEL, 1, rk_w, rk_b, DMODEL,
        Rp, (long long)N_COND*RELD, RELD, 1);

    // selector: l2norm rows -> column sums -> centrality -> exact top-k mask
    l2norm_kernel<<<(BATCH*N_COND + 255)/256, 256, 0, stream>>>(Rp, BATCH*N_COND, RELD);
    colsum_kernel<<<BATCH, RELD, 0, stream>>>(Rp, Csum);
    cent_kernel<<<(BATCH*N_COND + 255)/256, 256, 0, stream>>>(Rp, Csum, Cent);
    select_kernel<<<BATCH, N_COND, 0, stream>>>(Cent, mask, Mbias);

    // masked multi-head flash attention (all-f16 operands, f32 accum)
    flash_attn_kernel<<<dim3(S_TOK/16, NHEADS, BATCH), 32, 0, stream>>>(
        Qh, Kh, Vth, Mbias, AOh);

    // final projection from f16 attention output, written transposed into (B,C,S)
    gemm_abT_wmma<_Float16, float><<<dim3(S_TOK/16, DMODEL/16, BATCH), 32, 0, stream>>>(
        AOh, (long long)S_TOK*DMODEL, DMODEL, 1, o_w, o_b, DMODEL,
        out, (long long)DMODEL*S_TOK, /*oM*/1, /*oN*/S_TOK);
}
